// LiquidTransformerLayer_25795573580558
// MI455X (gfx1250) — matrix operand order repaired and checked by reading, not compiled, not run on hardware
//
#include <hip/hip_runtime.h>
#include <hip/hip_bf16.h>
#include <math.h>

typedef __attribute__((ext_vector_type(16))) _Float16 v16h;
typedef __attribute__((ext_vector_type(8)))  _Float16 v8h;
typedef __attribute__((ext_vector_type(8)))  float    v8f;
typedef __attribute__((ext_vector_type(8)))  unsigned int v8u;

#define TOK   2048
#define EMB   256
#define SEQ   512
#define NHEAD 8

// ---------------------------------------------------------------- utilities

__device__ __forceinline__ float gelu_f(float x) {
  // exact (erf) GELU, matches jax.nn.gelu(approximate=False)
  return 0.5f * x * (1.0f + erff(x * 0.70710678118654752440f));
}

// block-wide sum over exactly 256 threads (8 waves of 32)
__device__ __forceinline__ float block_sum_256(float v) {
  __shared__ float red[8];
  int lane = threadIdx.x & 31, wave = threadIdx.x >> 5;
#pragma unroll
  for (int off = 16; off > 0; off >>= 1) v += __shfl_xor(v, off, 32);
  if (lane == 0) red[wave] = v;
  __syncthreads();
  float tot = 0.f;
#pragma unroll
  for (int i = 0; i < 8; ++i) tot += red[i];
  __syncthreads();
  return tot;
}

// ---------------------------------------------------------------- WMMA GEMM
// C[M,N] (=|+=) optScale(row) * (A[M,K] @ B[K,N] + bias[N])
// A,B row-major f32 in global; staged to LDS as f16; f32 accumulate in WMMA.
// Block: 256 threads = 8 waves; block tile 128(M) x 64(N); K step 32.
// Wave grid 4(M) x 2(N); each wave computes a 32x32 tile = 4 WMMA per K-step
// with 2x2 fragment reuse. Requires M % 128 == 0, K % 32 == 0 (true for all
// GEMMs here); ragged N (router logits N=8) handled by a block-uniform path.

__global__ __launch_bounds__(256) void gemm_wmma_kernel(
    const float* __restrict__ A, const float* __restrict__ Bw,
    const float* __restrict__ bias, float* __restrict__ C,
    int M, int N, int K,
    const float* __restrict__ rowScale, int rsStride, int accumulate)
{
  // stride 40 halfs (80B): keeps 16B alignment for vector stores, non-pow2 banks
  __shared__ _Float16 As[128][40];    // A tile, row-major  [m][k]
  __shared__ _Float16 BsT[64][40];    // B tile, transposed [n][k]

  const int tid  = threadIdx.x;
  const int wave = tid >> 5;
  const int lane = tid & 31;
  const int wm = wave >> 1;            // 0..3 : 32-row band
  const int wn = wave & 1;             // 0..1 : 32-col band
  const int m16  = lane & 15;
  const int half = lane >> 4;

  const int blockM = blockIdx.x * 128;
  const int blockN = blockIdx.y * 64;
  const bool fullN = (blockN + 64 <= N);   // block-uniform

  // staging assignments
  const int arow = tid >> 1;            // 0..127
  const int aseg = (tid & 1) << 4;      // 0 / 16
  const int brow = tid >> 3;            // 0..31  (K)
  const int bseg = (tid & 7) << 3;      // 0..56  (N)

  v8f acc[4];
  { v8f z = {}; acc[0] = z; acc[1] = z; acc[2] = z; acc[3] = z; }

  for (int k0 = 0; k0 < K; k0 += 32) {
    // ---- stage A: 128x32, 16 contiguous f32 per thread -> 16 f16
    {
      const float* ap = A + (size_t)(blockM + arow) * K + k0 + aseg;
      float4 a0 = ((const float4*)ap)[0];
      float4 a1 = ((const float4*)ap)[1];
      float4 a2 = ((const float4*)ap)[2];
      float4 a3 = ((const float4*)ap)[3];
      v8h h0 = { (_Float16)a0.x, (_Float16)a0.y, (_Float16)a0.z, (_Float16)a0.w,
                 (_Float16)a1.x, (_Float16)a1.y, (_Float16)a1.z, (_Float16)a1.w };
      v8h h1 = { (_Float16)a2.x, (_Float16)a2.y, (_Float16)a2.z, (_Float16)a2.w,
                 (_Float16)a3.x, (_Float16)a3.y, (_Float16)a3.z, (_Float16)a3.w };
      *(v8h*)&As[arow][aseg]     = h0;
      *(v8h*)&As[arow][aseg + 8] = h1;
    }
    // ---- stage B: 32x64 -> transposed LDS [n][k]
    if (fullN) {
      const float* bp = Bw + (size_t)(k0 + brow) * N + blockN + bseg;
      float4 b0 = ((const float4*)bp)[0];
      float4 b1 = ((const float4*)bp)[1];
      BsT[bseg + 0][brow] = (_Float16)b0.x;
      BsT[bseg + 1][brow] = (_Float16)b0.y;
      BsT[bseg + 2][brow] = (_Float16)b0.z;
      BsT[bseg + 3][brow] = (_Float16)b0.w;
      BsT[bseg + 4][brow] = (_Float16)b1.x;
      BsT[bseg + 5][brow] = (_Float16)b1.y;
      BsT[bseg + 6][brow] = (_Float16)b1.z;
      BsT[bseg + 7][brow] = (_Float16)b1.w;
    } else {
      // branchless clamp+mask: loads always execute at a valid address
#pragma unroll
      for (int i = 0; i < 8; ++i) {
        int gn  = blockN + bseg + i;
        int gnc = gn < N ? gn : N - 1;
        float vb = Bw[(size_t)(k0 + brow) * N + gnc];
        BsT[bseg + i][brow] = (_Float16)(gn < N ? vb : 0.0f);
      }
    }
    // prefetch next K tile (global_prefetch_b8)
    if (k0 + 32 < K) {
      __builtin_prefetch(A + (size_t)(blockM + arow) * K + k0 + 32 + aseg, 0, 1);
      __builtin_prefetch(Bw + (size_t)(k0 + 32 + brow) * N + blockN + (fullN ? bseg : 0), 0, 1);
    }
    __syncthreads();

    // ---- gather fragments (all dword LDS reads) per ISA 7.12.2 layouts
    v8u au[2], bu[2];
#pragma unroll
    for (int mi = 0; mi < 2; ++mi) {
      int am = wm * 32 + mi * 16 + m16;
#pragma unroll
      for (int v = 0; v < 8; ++v) {
        // A 16x32 f16: VGPR v, lanes0-15: K=(v/4)*16+(v%4)*2 ; lanes16-31: +8
        int ka = ((v >> 2) << 4) + (half << 3) + ((v & 3) << 1);
        au[mi][v] = *(const unsigned int*)&As[am][ka];
      }
    }
#pragma unroll
    for (int ni = 0; ni < 2; ++ni) {
      int bn = wn * 32 + ni * 16 + m16;
#pragma unroll
      for (int v = 0; v < 8; ++v) {
        // B 32x16 f16: VGPR v, lanes0-15: K=(v/4)*16+(v%4)*2 ; lanes16-31: +8
        int kb = ((v >> 2) << 4) + (half << 3) + ((v & 3) << 1);
        bu[ni][v] = *(const unsigned int*)&BsT[bn][kb];
      }
    }
#pragma unroll
    for (int mi = 0; mi < 2; ++mi)
#pragma unroll
      for (int ni = 0; ni < 2; ++ni) {
        v16h af = __builtin_bit_cast(v16h, au[mi]);
        v16h bf = __builtin_bit_cast(v16h, bu[ni]);
        acc[mi * 2 + ni] = __builtin_amdgcn_wmma_f32_16x16x32_f16(
            /*neg_a=*/false, af, /*neg_b=*/false, bf,
            /*c_mod=*/(short)0, acc[mi * 2 + ni],
            /*reuse_a=*/false, /*reuse_b=*/false);
      }
    __syncthreads();
  }

  // ---- epilogue: C/D layout — lane n = lane&15, VGPR r -> M = r + 8*half
#pragma unroll
  for (int mi = 0; mi < 2; ++mi)
#pragma unroll
    for (int ni = 0; ni < 2; ++ni) {
      int cn = blockN + wn * 32 + ni * 16 + m16;
      v8f a = acc[mi * 2 + ni];
#pragma unroll
      for (int r = 0; r < 8; ++r) {
        int cm = blockM + wm * 32 + mi * 16 + r + 8 * half;
        if (cn < N) {
          float val = a[r] + (bias ? bias[cn] : 0.0f);
          if (rowScale) val *= rowScale[(size_t)cm * rsStride];
          if (accumulate) C[(size_t)cm * N + cn] += val;
          else            C[(size_t)cm * N + cn]  = val;
        }
      }
    }
}

// ---------------------------------------------------------------- small kernels

// te = gelu(0.5*te_w1 + te_b1) @ te_w2 + te_b2      [256]
__global__ __launch_bounds__(256) void te_kernel(
    const float* __restrict__ tw1, const float* __restrict__ tb1,
    const float* __restrict__ tw2, const float* __restrict__ tb2,
    float* __restrict__ te)
{
  __shared__ float t1[128];
  int e = threadIdx.x;
  if (e < 128) t1[e] = gelu_f(0.5f * tw1[e] + tb1[e]);
  __syncthreads();
  float acc = tb2[e];
  for (int i = 0; i < 128; ++i) acc += t1[i] * tw2[i * EMB + e];
  te[e] = acc;
}

// xa = x + rel_pos[s] + te
__global__ __launch_bounds__(256) void add_pos_te_kernel(
    const float* __restrict__ x, const float* __restrict__ rel,
    const float* __restrict__ te, float* __restrict__ xa)
{
  int t = blockIdx.x, e = threadIdx.x;
  int s = t & (SEQ - 1);
  xa[(size_t)t * EMB + e] = x[(size_t)t * EMB + e] + rel[(size_t)s * EMB + e] + te[e];
}

// attention core per (b,h): softmax(q k^T / sqrt(32)) v ; dh == 32 == wave32
__global__ __launch_bounds__(256) void attn_core_kernel(
    const float* __restrict__ q, const float* __restrict__ k,
    const float* __restrict__ v, float* __restrict__ o)
{
  int bh = blockIdx.x;
  int b = bh / NHEAD, h = bh % NHEAD;
  int wave = threadIdx.x >> 5, lane = threadIdx.x & 31;
  __shared__ float p_s[8][SEQ];
  __shared__ float q_s[8][32];
  const float scale = 0.17677669529663687f;   // 1/sqrt(32)

  for (int r = wave; r < SEQ; r += 8) {
    int t = b * SEQ + r;
    q_s[wave][lane] = q[(size_t)t * EMB + h * 32 + lane];
    __syncthreads();

    float sc[16];
    float mymax = -1e30f;
    for (int jj = 0; jj < 16; ++jj) {
      int j = jj * 32 + lane;
      const float* kp = k + (size_t)(b * SEQ + j) * EMB + h * 32;
      float acc = 0.f;
#pragma unroll 8
      for (int d = 0; d < 32; ++d) acc += q_s[wave][d] * kp[d];
      acc *= scale;
      sc[jj] = acc;
      mymax = fmaxf(mymax, acc);
    }
#pragma unroll
    for (int off = 16; off > 0; off >>= 1) mymax = fmaxf(mymax, __shfl_xor(mymax, off, 32));
    float mysum = 0.f;
    for (int jj = 0; jj < 16; ++jj) {
      float ev = expf(sc[jj] - mymax);
      p_s[wave][jj * 32 + lane] = ev;
      mysum += ev;
    }
#pragma unroll
    for (int off = 16; off > 0; off >>= 1) mysum += __shfl_xor(mysum, off, 32);
    float inv = 1.f / mysum;
    __syncthreads();

    float acc = 0.f;
    for (int j = 0; j < SEQ; ++j)
      acc += p_s[wave][j] * v[(size_t)(b * SEQ + j) * EMB + h * 32 + lane];
    o[(size_t)t * EMB + h * 32 + lane] = acc * inv;
    __syncthreads();
  }
}

// out = LN(a [+ r] [+ br(broadcast per batch)], g, be) over 256 cols; 1 token/block
__global__ __launch_bounds__(256) void resid_ln_kernel(
    const float* __restrict__ a, const float* __restrict__ r,
    const float* __restrict__ br, const float* __restrict__ g,
    const float* __restrict__ be, float* __restrict__ out)
{
  int t = blockIdx.x, e = threadIdx.x;
  int b = t >> 9;
  float v = a[(size_t)t * EMB + e];
  if (r)  v += r[(size_t)t * EMB + e];
  if (br) v += br[b * EMB + e];
  float mean = block_sum_256(v) * (1.0f / EMB);
  float d = v - mean;
  float var = block_sum_256(d * d) * (1.0f / EMB);
  out[(size_t)t * EMB + e] = d * rsqrtf(var + 1e-5f) * g[e] + be[e];
}

// sr[b,e] = mean over s of x1
__global__ __launch_bounds__(256) void seq_mean_kernel(
    const float* __restrict__ x1, float* __restrict__ sr)
{
  int b = blockIdx.x, e = threadIdx.x;
  float acc = 0.f;
  for (int s = 0; s < SEQ; ++s) acc += x1[(size_t)(b * SEQ + s) * EMB + e];
  sr[b * EMB + e] = acc * (1.0f / SEQ);
}

// gate: n = 5 + floor(mean(sigmoid(sr@cg_w+cg_b)) * ode_steps); meta = {n, 1/(n-1)}
__global__ __launch_bounds__(256) void cell_gate_kernel(
    const float* __restrict__ sr, const float* __restrict__ cgw,
    const float* __restrict__ cgb, float ode_steps, float* __restrict__ meta)
{
  int e = threadIdx.x;
  float local = 0.f;
  for (int b = 0; b < 4; ++b) {
    float acc = cgb[e];
    for (int k = 0; k < 256; ++k) acc += sr[b * 256 + k] * cgw[k * 256 + e];
    local += 1.0f / (1.0f + expf(-acc));
  }
  float tot = block_sum_256(local);
  if (e == 0) {
    float cm = tot * (1.0f / 1024.0f);
    float n = 5.0f + floorf(cm * ode_steps);
    meta[0] = n;
    meta[1] = 1.0f / (n - 1.0f);
  }
}

// liquid cell Euler loop; one block per batch row
__global__ __launch_bounds__(256) void cell_run_kernel(
    const float* __restrict__ sr, const float* __restrict__ h0,
    const float* __restrict__ w1, const float* __restrict__ b1,
    const float* __restrict__ g1, const float* __restrict__ be1,
    const float* __restrict__ w2, const float* __restrict__ b2,
    const float* __restrict__ w3, const float* __restrict__ b3,
    const float* __restrict__ tmw, const float* __restrict__ tmb,
    const float* __restrict__ lng, const float* __restrict__ lnb,
    const float* __restrict__ meta, float inv_tc, int iters,
    float* __restrict__ hout)
{
  int b = blockIdx.x;
  int e = threadIdx.x;
  __shared__ float hsh[EMB];
  __shared__ float z1[1024];
  __shared__ float z2[512];
  hsh[e] = h0[b * EMB + e];
  float nf = meta[0];
  float dt = meta[1];
  const float* srb = sr + b * EMB;
  __syncthreads();

  for (int it = 0; it < iters; ++it) {
    float t = (float)it * dt;
    float zz[4];
#pragma unroll
    for (int qq = 0; qq < 4; ++qq) {
      int j = e + 256 * qq;
      float acc = b1[j];
      for (int k2 = 0; k2 < 256; ++k2) acc += hsh[k2] * w1[(size_t)k2 * 1024 + j];
      for (int k2 = 0; k2 < 256; ++k2) acc += srb[k2] * w1[(size_t)(256 + k2) * 1024 + j];
      zz[qq] = acc;
    }
    float mean = block_sum_256(zz[0] + zz[1] + zz[2] + zz[3]) * (1.0f / 1024.0f);
    float s2 = 0.f;
#pragma unroll
    for (int qq = 0; qq < 4; ++qq) { float d = zz[qq] - mean; s2 += d * d; }
    float inv = rsqrtf(block_sum_256(s2) * (1.0f / 1024.0f) + 1e-5f);
#pragma unroll
    for (int qq = 0; qq < 4; ++qq) {
      int j = e + 256 * qq;
      z1[j] = gelu_f((zz[qq] - mean) * inv * g1[j] + be1[j]);
    }
    __syncthreads();
#pragma unroll
    for (int qq = 0; qq < 2; ++qq) {
      int j = e + 256 * qq;
      float acc = b2[j];
      for (int k2 = 0; k2 < 1024; ++k2) acc += z1[k2] * w2[(size_t)k2 * 512 + j];
      z2[j] = gelu_f(acc);
    }
    __syncthreads();
    float acc = b3[e];
    for (int k2 = 0; k2 < 512; ++k2) acc += z2[k2] * w3[(size_t)k2 * 256 + e];
    float dh = acc * tanhf(t * tmw[e] + tmb[e]) * inv_tc;
    float hn = ((float)it < nf - 1.0f) ? (hsh[e] + dt * dh) : hsh[e];
    hsh[e] = hn;
    __syncthreads();
  }
  // final LN
  float hv = hsh[e];
  float mean = block_sum_256(hv) * (1.0f / 256.0f);
  float d = hv - mean;
  float var = block_sum_256(d * d) * (1.0f / 256.0f);
  hout[b * EMB + e] = d * rsqrtf(var + 1e-5f) * lng[e] + lnb[e];
}

// softmax mix of the 3 cell states
__global__ __launch_bounds__(256) void mix_kernel(
    const float* __restrict__ sr, const float* __restrict__ mw,
    const float* __restrict__ mb, const float* __restrict__ fh,
    const float* __restrict__ mh, const float* __restrict__ sh,
    float* __restrict__ liquid)
{
  int b = blockIdx.x, e = threadIdx.x;
  __shared__ float wsm[3];
  if (e < 3) {
    float acc = mb[e];
    for (int k2 = 0; k2 < 256; ++k2) acc += sr[b * 256 + k2] * mw[k2 * 3 + e];
    wsm[e] = acc;
  }
  __syncthreads();
  if (e == 0) {
    float m = fmaxf(wsm[0], fmaxf(wsm[1], wsm[2]));
    float e0 = expf(wsm[0] - m), e1 = expf(wsm[1] - m), e2 = expf(wsm[2] - m);
    float s = e0 + e1 + e2;
    wsm[0] = e0 / s; wsm[1] = e1 / s; wsm[2] = e2 / s;
  }
  __syncthreads();
  liquid[b * 256 + e] = wsm[0] * fh[b * 256 + e] + wsm[1] * mh[b * 256 + e]
                      + wsm[2] * sh[b * 256 + e];
}

// cc[t] = [x2[t], liquid[b]]
__global__ __launch_bounds__(256) void concat_kernel(
    const float* __restrict__ x2, const float* __restrict__ liquid,
    float* __restrict__ cc)
{
  int t = blockIdx.x, e = threadIdx.x;
  int b = t >> 9;
  cc[(size_t)t * 512 + e]       = x2[(size_t)t * EMB + e];
  cc[(size_t)t * 512 + 256 + e] = liquid[b * EMB + e];
}

__global__ void gelu_kernel(float* y, int n) {
  int i = blockIdx.x * 256 + threadIdx.x;
  if (i < n) y[i] = gelu_f(y[i]);
}
__global__ void zero_kernel(float* y, int n) {
  int i = blockIdx.x * 256 + threadIdx.x;
  if (i < n) y[i] = 0.f;
}

// per-token LN(+gamma/beta)+GELU over variable width W (expert hidden)
__global__ __launch_bounds__(256) void ln_gelu_rows_kernel(
    float* __restrict__ z, const float* __restrict__ g,
    const float* __restrict__ be, int W)
{
  size_t base = (size_t)blockIdx.x * (size_t)W;
  float s1 = 0.f;
  for (int j = threadIdx.x; j < W; j += 256) s1 += z[base + j];
  float mean = block_sum_256(s1) / (float)W;
  float s2 = 0.f;
  for (int j = threadIdx.x; j < W; j += 256) { float d = z[base + j] - mean; s2 += d * d; }
  float inv = rsqrtf(block_sum_256(s2) / (float)W + 1e-5f);
  for (int j = threadIdx.x; j < W; j += 256) {
    float val = (z[base + j] - mean) * inv * g[j] + be[j];
    z[base + j] = gelu_f(val);
  }
}

// softmax over 8 logits, top-2, renormalize -> per-expert weights [TOK,8]
__global__ __launch_bounds__(256) void router_topk_kernel(
    const float* __restrict__ logits, float* __restrict__ wbuf)
{
  int t = blockIdx.x * 256 + threadIdx.x;
  float lg[8];
  float mx = -1e30f;
#pragma unroll
  for (int e = 0; e < 8; ++e) { lg[e] = logits[t * 8 + e]; mx = fmaxf(mx, lg[e]); }
  float s = 0.f;
#pragma unroll
  for (int e = 0; e < 8; ++e) { lg[e] = expf(lg[e] - mx); s += lg[e]; }
  float invs = 1.f / s;
#pragma unroll
  for (int e = 0; e < 8; ++e) lg[e] *= invs;
  int i0 = 0;
#pragma unroll
  for (int e = 1; e < 8; ++e) if (lg[e] > lg[i0]) i0 = e;
  int i1 = (i0 == 0) ? 1 : 0;
#pragma unroll
  for (int e = 0; e < 8; ++e) if (e != i0 && lg[e] > lg[i1]) i1 = e;
  float den = lg[i0] + lg[i1];
#pragma unroll
  for (int e = 0; e < 8; ++e)
    wbuf[t * 8 + e] = (e == i0) ? lg[i0] / den : ((e == i1) ? lg[i1] / den : 0.f);
}

// ---------------------------------------------------------------- launcher

extern "C" void kernel_launch(void* const* d_in, const int* in_sizes, int n_in,
                              void* d_out, int out_size, void* d_ws, size_t ws_size,
                              hipStream_t stream) {
  (void)in_sizes; (void)n_in; (void)out_size; (void)ws_size;

  // ------- inputs (setup_inputs dict order, flattened) -------
  const float* x      = (const float*)d_in[0];
  const float* fast_h = (const float*)d_in[1];
  const float* med_h  = (const float*)d_in[2];
  const float* slow_h = (const float*)d_in[3];
  const float* rel_pos= (const float*)d_in[4];
  const float* te_w1  = (const float*)d_in[5];
  const float* te_b1  = (const float*)d_in[6];
  const float* te_w2  = (const float*)d_in[7];
  const float* te_b2  = (const float*)d_in[8];
  const float* wq = (const float*)d_in[9];   const float* bq = (const float*)d_in[10];
  const float* wk = (const float*)d_in[11];  const float* bk = (const float*)d_in[12];
  const float* wv = (const float*)d_in[13];  const float* bv = (const float*)d_in[14];
  const float* wo = (const float*)d_in[15];  const float* bo = (const float*)d_in[16];
  // cells: fast @17, medium @31, slow @45 (14 tensors each)
  const float* mix_w = (const float*)d_in[59];
  const float* mix_b = (const float*)d_in[60];
  const float* rw1 = (const float*)d_in[61]; const float* rb1 = (const float*)d_in[62];
  const float* rw2 = (const float*)d_in[63]; const float* rb2 = (const float*)d_in[64];
  // experts @65 (6 tensors each, 8 experts)
  const float* n1g = (const float*)d_in[113]; const float* n1b = (const float*)d_in[114];
  const float* n2g = (const float*)d_in[115]; const float* n2b = (const float*)d_in[116];
  const float* n3g = (const float*)d_in[117]; const float* n3b = (const float*)d_in[118];

  // ------- workspace layout (floats) -------
  float* ws = (float*)d_ws;
  size_t off = 0;
  auto wsAlloc = [&](size_t n) { float* p = ws + off; off += n; return p; };
  float* te     = wsAlloc(256);
  float* xa     = wsAlloc((size_t)TOK * EMB);
  float* qb     = wsAlloc((size_t)TOK * EMB);
  float* kb     = wsAlloc((size_t)TOK * EMB);
  float* vb     = wsAlloc((size_t)TOK * EMB);
  float* attno  = wsAlloc((size_t)TOK * EMB);
  float* proj   = wsAlloc((size_t)TOK * EMB);
  float* x1b    = wsAlloc((size_t)TOK * EMB);
  float* sr     = wsAlloc(4 * EMB);
  float* metaF  = wsAlloc(8);
  float* fh     = wsAlloc(4 * EMB);
  float* mh     = wsAlloc(4 * EMB);
  float* shh    = wsAlloc(4 * EMB);
  float* liquid = wsAlloc(4 * EMB);
  float* x2b    = wsAlloc((size_t)TOK * EMB);
  float* cc     = wsAlloc((size_t)TOK * 512);
  float* rh     = wsAlloc((size_t)TOK * 1024);
  float* logits = wsAlloc((size_t)TOK * 8);
  float* wbuf   = wsAlloc((size_t)TOK * 8);
  float* moeacc = wsAlloc((size_t)TOK * EMB);
  float* zbuf   = wsAlloc((size_t)TOK * 8192);   // max expert hidden

  // ------- attention -------
  te_kernel<<<1, 256, 0, stream>>>(te_w1, te_b1, te_w2, te_b2, te);
  add_pos_te_kernel<<<TOK, 256, 0, stream>>>(x, rel_pos, te, xa);
  dim3 gP(TOK / 128, EMB / 64);
  gemm_wmma_kernel<<<gP, 256, 0, stream>>>(xa, wq, bq, qb, TOK, EMB, EMB, nullptr, 0, 0);
  gemm_wmma_kernel<<<gP, 256, 0, stream>>>(xa, wk, bk, kb, TOK, EMB, EMB, nullptr, 0, 0);
  gemm_wmma_kernel<<<gP, 256, 0, stream>>>(xa, wv, bv, vb, TOK, EMB, EMB, nullptr, 0, 0);
  attn_core_kernel<<<4 * NHEAD, 256, 0, stream>>>(qb, kb, vb, attno);
  gemm_wmma_kernel<<<gP, 256, 0, stream>>>(attno, wo, bo, proj, TOK, EMB, EMB, nullptr, 0, 0);
  resid_ln_kernel<<<TOK, 256, 0, stream>>>(x, proj, nullptr, n1g, n1b, x1b);

  // ------- liquid cells -------
  seq_mean_kernel<<<4, 256, 0, stream>>>(x1b, sr);
  const float* hins[3]  = {fast_h, med_h, slow_h};
  float        odes[3]  = {5.f, 10.f, 15.f};
  float        invtc[3] = {10.f, 1.f, 0.1f};
  int          iters[3] = {9, 14, 19};          // 5 + ode_steps - 1
  float*       houts[3] = {fh, mh, shh};
  for (int c = 0; c < 3; ++c) {
    int base = 17 + 14 * c;
    // order: w1,b1,g1,be1,w2,b2,w3,b3,cg_w,cg_b,tm_w,tm_b,ln_g,ln_b
    const float* cw1 = (const float*)d_in[base + 0];
    const float* cb1 = (const float*)d_in[base + 1];
    const float* cg1 = (const float*)d_in[base + 2];
    const float* cbe1= (const float*)d_in[base + 3];
    const float* cw2 = (const float*)d_in[base + 4];
    const float* cb2 = (const float*)d_in[base + 5];
    const float* cw3 = (const float*)d_in[base + 6];
    const float* cb3 = (const float*)d_in[base + 7];
    const float* cgw = (const float*)d_in[base + 8];
    const float* cgb = (const float*)d_in[base + 9];
    const float* ctmw= (const float*)d_in[base + 10];
    const float* ctmb= (const float*)d_in[base + 11];
    const float* clng= (const float*)d_in[base + 12];
    const float* clnb= (const float*)d_in[base + 13];
    cell_gate_kernel<<<1, 256, 0, stream>>>(sr, cgw, cgb, odes[c], metaF + 2 * c);
    cell_run_kernel<<<4, 256, 0, stream>>>(sr, hins[c],
        cw1, cb1, cg1, cbe1, cw2, cb2, cw3, cb3, ctmw, ctmb, clng, clnb,
        metaF + 2 * c, invtc[c], iters[c], houts[c]);
  }
  mix_kernel<<<4, 256, 0, stream>>>(sr, mix_w, mix_b, fh, mh, shh, liquid);
  resid_ln_kernel<<<TOK, 256, 0, stream>>>(x1b, nullptr, liquid, n2g, n2b, x2b);

  // ------- MoE router -------
  concat_kernel<<<TOK, 256, 0, stream>>>(x2b, liquid, cc);
  gemm_wmma_kernel<<<dim3(TOK / 128, 1024 / 64), 256, 0, stream>>>(
      cc, rw1, rb1, rh, TOK, 1024, 512, nullptr, 0, 0);
  gelu_kernel<<<(TOK * 1024) / 256, 256, 0, stream>>>(rh, TOK * 1024);
  gemm_wmma_kernel<<<dim3(TOK / 128, 1), 256, 0, stream>>>(
      rh, rw2, rb2, logits, TOK, 8, 1024, nullptr, 0, 0);
  router_topk_kernel<<<TOK / 256, 256, 0, stream>>>(logits, wbuf);

  // ------- MoE experts (sequential, reusing zbuf) -------
  zero_kernel<<<(TOK * EMB) / 256, 256, 0, stream>>>(moeacc, TOK * EMB);
  for (int e = 0; e < 8; ++e) {
    int he = 1024 * (e + 1);
    const float* ew1 = (const float*)d_in[65 + 6 * e + 0];
    const float* eb1 = (const float*)d_in[65 + 6 * e + 1];
    const float* eg  = (const float*)d_in[65 + 6 * e + 2];
    const float* ebe = (const float*)d_in[65 + 6 * e + 3];
    const float* ew2 = (const float*)d_in[65 + 6 * e + 4];
    const float* eb2 = (const float*)d_in[65 + 6 * e + 5];
    gemm_wmma_kernel<<<dim3(TOK / 128, he / 64), 256, 0, stream>>>(
        x2b, ew1, eb1, zbuf, TOK, he, EMB, nullptr, 0, 0);
    ln_gelu_rows_kernel<<<TOK, 256, 0, stream>>>(zbuf, eg, ebe, he);
    gemm_wmma_kernel<<<dim3(TOK / 128, EMB / 64), 256, 0, stream>>>(
        zbuf, ew2, eb2, moeacc, TOK, EMB, he, wbuf + e, 8, 1);
  }

  // ------- final residual LN -> output -------
  resid_ln_kernel<<<TOK, 256, 0, stream>>>(x2b, moeacc, nullptr, n3g, n3b, (float*)d_out);
}
